// DraftDecoderBlock_6803228197395
// MI455X (gfx1250) — matrix operand order repaired and checked by reading, not compiled, not run on hardware
//
#include <hip/hip_runtime.h>
#include <hip/hip_bf16.h>

// ---------------------------------------------------------------------------
// DraftDecoderBlock for MI455X (gfx1250): all GEMMs on v_wmma_f32_16x16x32_bf16
// Fragment layouts arranged so every WMMA operand is built from contiguous
// 16-byte LDS/global loads (ds_load_b128 / global_load_b128) + v_cvt_pk_bf16.
// ---------------------------------------------------------------------------

typedef __attribute__((ext_vector_type(16))) __bf16 v16bf;
typedef __attribute__((ext_vector_type(8)))  __bf16 v8bf;
typedef __attribute__((ext_vector_type(8)))  float  v8f;
typedef __attribute__((ext_vector_type(4)))  float  f4;

#define B_   4
#define T_   1024
#define S_   1024
#define DM   512
#define H_   8
#define DI   1024
#define CONVD 1152     // D_INNER + 2*D_STATE
#define DPROJ 2192     // 2*D_INNER + 2*D_STATE + SSM_H
#define BT   4096
#define SSH  16

// workspace layout (float offsets)
#define OFF_XN 0                              // 2,097,152  (ln outputs, reused 3x)
#define OFF_B  2097152                        // 8,978,432  (zxbcdt / q_raw+kv_raw / ffn_h)
#define OFF_C  11075584                       // 4,718,592  (xconv / q_rope + k_rope)
#define OFF_D  15794176                       // 4,194,304  (ssm_y / v^T + attn_out)
#define OFF_E  19988480                       // 2,097,152  (x after mamba)
#define OFF_F  22085632                       // 2,097,152  (x after attention)
#define OFF_G  24182784                       // 65,536     (softmax row max / row sum)
#define OFF_H  24248320                       // 4,096      (coverage accumulator)

__device__ __forceinline__ v8f wmma_bf16(v16bf a, v16bf b, v8f c) {
  return __builtin_amdgcn_wmma_f32_16x16x32_bf16(false, a, false, b, (short)0, c, false, false);
}
__device__ __forceinline__ v8f zero8() { v8f z = {0.f,0.f,0.f,0.f,0.f,0.f,0.f,0.f}; return z; }

// 8 consecutive f32 -> 8 bf16 (two 16B loads + packed converts)
__device__ __forceinline__ v8bf cvt8(const float* p) {
  f4 a = *(const f4*)p;
  f4 b = *(const f4*)(p + 4);
  v8bf r;
  r[0]=(__bf16)a[0]; r[1]=(__bf16)a[1]; r[2]=(__bf16)a[2]; r[3]=(__bf16)a[3];
  r[4]=(__bf16)b[0]; r[5]=(__bf16)b[1]; r[6]=(__bf16)b[2]; r[7]=(__bf16)b[3];
  return r;
}
__device__ __forceinline__ v16bf cat16(v8bf lo, v8bf hi) {
  return __builtin_shufflevector(lo, hi, 0,1,2,3,4,5,6,7,8,9,10,11,12,13,14,15);
}
// 16 consecutive f32 -> v16bf
__device__ __forceinline__ v16bf cvt16(const float* p) { return cat16(cvt8(p), cvt8(p + 8)); }

// A-fragment (16x32 MxK): per lane = row m,
//   K in [lh*16, lh*16+16)
__device__ __forceinline__ v16bf fragA_lds(const __bf16* row, int lh) {
  return cat16(*(const v8bf*)(row + lh * 16), *(const v8bf*)(row + lh * 16 + 8));
}
// B-fragment (32x16 KxN) from K-contiguous (transposed) storage:
//   per lane = col n row, K in [lh*16, lh*16+16)
__device__ __forceinline__ v16bf fragB_lds(const __bf16* row, int lh) {
  return cat16(*(const v8bf*)(row + lh * 16), *(const v8bf*)(row + lh * 16 + 8));
}

// ---------------------------------------------------------------------------
// GEMM: C[M,N] = act(A[M,K] @ W[N,K]^T + bias) + resid
// 256 threads (8 waves), block tile 64x64, K-step 32.
// LDS: A row-major bf16, B stored K-contiguous per n (i.e. W rows verbatim).
// ---------------------------------------------------------------------------
__global__ __launch_bounds__(256)
void gemm_bf16_wmma(const float* __restrict__ A, const float* __restrict__ W,
                    const float* __restrict__ bias, const float* __restrict__ resid,
                    float* __restrict__ C, int M, int N, int K, int act)
{
  __builtin_amdgcn_s_wait_tensorcnt(0);   // CDNA5 counter probe (NOP when idle)
  __shared__ alignas(16) __bf16 lA [64][40];   // rows 80B -> 16B aligned
  __shared__ alignas(16) __bf16 lBt[64][40];   // lBt[n][k]
  const int m0 = blockIdx.y * 64;
  const int n0 = blockIdx.x * 64;
  const int tid = threadIdx.x;
  const int wave = tid >> 5, lane = tid & 31;
  const int l16 = lane & 15, lh = lane >> 4;
  const int wr = wave >> 1;               // wave tile row 0..3
  const int wc = (wave & 1) * 2;          // wave tile col base 0 or 2

  // staging: each thread moves 8 contiguous floats of A and of W per K-step
  const int sr  = tid >> 2;               // 0..63
  const int sc0 = (tid & 3) * 8;          // 0,8,16,24
  const float* Aptr = A + (size_t)(m0 + sr) * K + sc0;
  const float* Wptr = W + (size_t)(n0 + sr) * K + sc0;
  const bool wvalid = (n0 + sr) < N;      // N edge (in_proj N=2192)

  v8f acc0 = zero8(), acc1 = zero8();

  for (int k0 = 0; k0 < K; k0 += 32) {
    *(v8bf*)&lA[sr][sc0] = cvt8(Aptr);
    if (wvalid) {
      *(v8bf*)&lBt[sr][sc0] = cvt8(Wptr);
    } else {
      v8bf z; 
      #pragma unroll
      for (int e = 0; e < 8; e++) z[e] = (__bf16)0.f;
      *(v8bf*)&lBt[sr][sc0] = z;
    }
    Aptr += 32; Wptr += 32;
    if (k0 + 32 < K) { __builtin_prefetch(Aptr, 0, 1); __builtin_prefetch(Wptr, 0, 1); }
    __syncthreads();

    v16bf af = fragA_lds(&lA[wr * 16 + l16][0], lh);
    v16bf b0 = fragB_lds(&lBt[(wc + 0) * 16 + l16][0], lh);
    v16bf b1 = fragB_lds(&lBt[(wc + 1) * 16 + l16][0], lh);
    acc0 = wmma_bf16(af, b0, acc0);
    acc1 = wmma_bf16(af, b1, acc1);
    __syncthreads();
  }

  #pragma unroll
  for (int t = 0; t < 2; t++) {
    v8f acc = t ? acc1 : acc0;
    int n = n0 + (wc + t) * 16 + l16;
    if (n < N) {
      float bv = bias ? bias[n] : 0.f;
      #pragma unroll
      for (int r = 0; r < 8; r++) {
        int m = m0 + wr * 16 + r + (lh << 3);
        float v = acc[r] + bv;
        if (act == 1) v = 0.5f * v * (1.0f + erff(v * 0.70710678118f));  // exact GELU
        if (resid) v += resid[(size_t)m * N + n];
        C[(size_t)m * N + n] = v;
      }
    }
  }
}

// ---------------------------------------------------------------------------
// LayerNorm over last dim (512), one row per block
// ---------------------------------------------------------------------------
__global__ __launch_bounds__(256)
void layernorm_k(const float* __restrict__ x, const float* __restrict__ w,
                 const float* __restrict__ b, float* __restrict__ y)
{
  const int row = blockIdx.x, tid = threadIdx.x;
  const float* xr = x + (size_t)row * DM;
  float a0 = xr[tid], a1 = xr[tid + 256];
  __shared__ float red[256];
  red[tid] = a0 + a1; __syncthreads();
  for (int s = 128; s > 0; s >>= 1) { if (tid < s) red[tid] += red[tid + s]; __syncthreads(); }
  float mean = red[0] * (1.0f / DM);
  __syncthreads();
  float d0 = a0 - mean, d1 = a1 - mean;
  red[tid] = d0 * d0 + d1 * d1; __syncthreads();
  for (int s = 128; s > 0; s >>= 1) { if (tid < s) red[tid] += red[tid + s]; __syncthreads(); }
  float rstd = rsqrtf(red[0] * (1.0f / DM) + 1e-5f);
  y[(size_t)row * DM + tid]       = d0 * rstd * w[tid]       + b[tid];
  y[(size_t)row * DM + tid + 256] = d1 * rstd * w[tid + 256] + b[tid + 256];
}

// ---------------------------------------------------------------------------
// Causal depthwise conv (k=4) + SiLU.  xBC lives in zxbcdt[:, 1024:2176)
// ---------------------------------------------------------------------------
__global__ __launch_bounds__(256)
void conv_silu_k(const float* __restrict__ zx, const float* __restrict__ cw,
                 const float* __restrict__ cb, float* __restrict__ out)
{
  const int idx = blockIdx.x * 256 + threadIdx.x;           // B*T*1152
  const int c  = idx % CONVD;
  const int bt = idx / CONVD;
  const int t  = bt & 1023, b = bt >> 10;
  float acc = cb[c];
  #pragma unroll
  for (int j = 0; j < 4; j++) {
    int tt = t - 3 + j;
    if (tt >= 0) acc += cw[c * 4 + j] * zx[((size_t)(b * T_ + tt)) * DPROJ + DI + c];
  }
  out[idx] = acc / (1.0f + __expf(-acc));                   // silu
}

// ---------------------------------------------------------------------------
// SSM scan: one block per (b,h), state h[64x64] striped over 256 threads
// ---------------------------------------------------------------------------
__global__ __launch_bounds__(256)
void ssm_scan_k(const float* __restrict__ zx, const float* __restrict__ xc,
                const float* __restrict__ dt_bias, const float* __restrict__ A_log,
                const float* __restrict__ Dp, float* __restrict__ y)
{
  const int bh = blockIdx.x;
  const int b = bh >> 4, h = bh & 15;
  const int tid = threadIdx.x;
  const int p = tid & 63, ng = tid >> 6;
  __shared__ float sB[64], sC[64], sRed[4][64];
  float hs[16];
  #pragma unroll
  for (int i = 0; i < 16; i++) hs[i] = 0.f;
  const float Acoef = -__expf(A_log[h]);
  const float dbias = dt_bias[h];
  const float Dh = Dp[h];

  for (int t = 0; t < T_; t++) {
    const float* xrow = xc + (size_t)(b * T_ + t) * CONVD;
    if (tid < 64)       sB[tid]      = xrow[DI + tid];
    else if (tid < 128) sC[tid - 64] = xrow[DI + 64 + (tid - 64)];
    float raw = zx[(size_t)(b * T_ + t) * DPROJ + (DPROJ - SSH) + h] + dbias;
    float dt  = (raw > 20.f) ? raw : log1pf(__expf(raw));
    float dA  = __expf(dt * Acoef);
    float xp  = xrow[h * 64 + p];
    float dtx = dt * xp;
    __syncthreads();
    float part = 0.f;
    #pragma unroll
    for (int i = 0; i < 16; i++) {
      int n = ng * 16 + i;
      hs[i] = hs[i] * dA + dtx * sB[n];
      part += hs[i] * sC[n];
    }
    sRed[ng][p] = part;
    __syncthreads();
    if (ng == 0)
      y[(size_t)(b * T_ + t) * DI + h * 64 + p] =
          sRed[0][p] + sRed[1][p] + sRed[2][p] + sRed[3][p] + Dh * xp;
    __syncthreads();
  }
}

// ---------------------------------------------------------------------------
// y = rmsnorm(y * silu(z)) * w   (in place, 1024 wide)
// ---------------------------------------------------------------------------
__global__ __launch_bounds__(256)
void gate_rmsnorm_k(float* __restrict__ y, const float* __restrict__ zx,
                    const float* __restrict__ nw)
{
  const int row = blockIdx.x, tid = threadIdx.x;
  float g[4]; float ss = 0.f;
  #pragma unroll
  for (int i = 0; i < 4; i++) {
    int c = tid + i * 256;
    float z = zx[(size_t)row * DPROJ + c];
    float v = y[(size_t)row * DI + c] * (z / (1.0f + __expf(-z)));
    g[i] = v; ss += v * v;
  }
  __shared__ float red[256];
  red[tid] = ss; __syncthreads();
  for (int s = 128; s > 0; s >>= 1) { if (tid < s) red[tid] += red[tid + s]; __syncthreads(); }
  float sc = rsqrtf(red[0] * (1.0f / DI) + 1e-5f);
  #pragma unroll
  for (int i = 0; i < 4; i++) {
    int c = tid + i * 256;
    y[(size_t)row * DI + c] = g[i] * sc * nw[c];
  }
}

// ---------------------------------------------------------------------------
// RoPE + head-major repack. q scaled by 1/sqrt(64)=0.125.
// V is stored TRANSPOSED: vt[(bh*64 + d)*1024 + s] so P@V B-fragments are
// contiguous 16-float loads.
// ---------------------------------------------------------------------------
__global__ __launch_bounds__(256)
void rope_q_k(const float* __restrict__ qraw, float* __restrict__ qr, const int* __restrict__ qoff)
{
  const int idx = blockIdx.x * 256 + threadIdx.x;           // B*T*512
  const int d = idx & 63, h = (idx >> 6) & 7, t = (idx >> 9) & 1023, b = idx >> 19;
  const int i = d & 31;
  float inv = __expf(-(float)i * (9.210340372f / 32.f));    // 10000^{-i/32}
  float ang = (float)(t + *qoff) * inv;
  const float* row = qraw + (size_t)(b * T_ + t) * DM + h * 64;
  float v = row[d];
  float rot = (d < 32) ? -row[d + 32] : row[d - 32];
  qr[((size_t)(b * H_ + h) * T_ + t) * 64 + d] = 0.125f * (v * __cosf(ang) + rot * __sinf(ang));
}

__global__ __launch_bounds__(256)
void rope_kv_k(const float* __restrict__ kvraw, float* __restrict__ kr, float* __restrict__ vt)
{
  const int idx = blockIdx.x * 256 + threadIdx.x;           // B*S*512
  const int d = idx & 63, h = (idx >> 6) & 7, s = (idx >> 9) & 1023, b = idx >> 19;
  const int i = d & 31;
  float inv = __expf(-(float)i * (9.210340372f / 32.f));
  float ang = (float)s * inv;
  const int bh = b * H_ + h;
  const float* rowk = kvraw + (size_t)(b * S_ + s) * (2 * DM) + h * 64;
  float v = rowk[d];
  float rot = (d < 32) ? -rowk[d + 32] : rowk[d - 32];
  kr[((size_t)bh * S_ + s) * 64 + d] = v * __cosf(ang) + rot * __sinf(ang);
  vt[((size_t)bh * 64 + d) * S_ + s] =
      kvraw[(size_t)(b * S_ + s) * (2 * DM) + DM + h * 64 + d];
}

// ---------------------------------------------------------------------------
// Attention pass 1: row max & sumexp per (b,h,t). One wave per 16 t-rows.
// ---------------------------------------------------------------------------
__global__ __launch_bounds__(32)
void attn_pass1(const float* __restrict__ qr, const float* __restrict__ kr,
                float* __restrict__ mstat, float* __restrict__ lstat)
{
  const int blk = blockIdx.x;                               // B*H*(T/16)
  const int tt = blk & 63, bh = blk >> 6;
  const float* qb = qr + ((size_t)bh * T_ + tt * 16) * 64;
  const float* kb = kr + (size_t)bh * S_ * 64;
  const int lane = threadIdx.x, l16 = lane & 15, lh = lane >> 4;
  __shared__ alignas(16) float sc[16][36];

  // Q fragments: 16 contiguous floats per lane (K in [lh*16, lh*16+16)), per 32-wide K chunk
  v16bf aq0 = cvt16(qb + l16 * 64 + lh * 16);
  v16bf aq1 = cvt16(qb + l16 * 64 + 32 + lh * 16);
  float mrun = -1e30f, lrun = 0.f;

  for (int s0 = 0; s0 < S_; s0 += 32) {
    #pragma unroll
    for (int sn = 0; sn < 2; sn++) {
      const float* krow = kb + (size_t)(s0 + sn * 16 + l16) * 64;
      v8f acc = zero8();
      acc = wmma_bf16(aq0, cvt16(krow + lh * 16), acc);        // d 0..31
      acc = wmma_bf16(aq1, cvt16(krow + 32 + lh * 16), acc);   // d 32..63
      #pragma unroll
      for (int r = 0; r < 8; r++) sc[r + lh * 8][sn * 16 + l16] = acc[r];
    }
    __syncthreads();
    if (lane < 16) {
      float cmax = mrun;
      for (int c = 0; c < 32; c++) cmax = fmaxf(cmax, sc[lane][c]);
      float ssum = 0.f;
      for (int c = 0; c < 32; c++) ssum += __expf(sc[lane][c] - cmax);
      lrun = lrun * __expf(mrun - cmax) + ssum;
      mrun = cmax;
    }
    __syncthreads();
  }
  if (lane < 16) {
    mstat[(size_t)bh * T_ + tt * 16 + lane] = mrun;
    lstat[(size_t)bh * T_ + tt * 16 + lane] = lrun;
  }
}

// ---------------------------------------------------------------------------
// Attention pass 2: w = exp(score-m)/l, O = w @ V (WMMA), coverage += colsums
// ---------------------------------------------------------------------------
__global__ __launch_bounds__(32)
void attn_pass2(const float* __restrict__ qr, const float* __restrict__ kr,
                const float* __restrict__ vt, const float* __restrict__ mstat,
                const float* __restrict__ lstat, float* __restrict__ attn,
                float* __restrict__ cov)
{
  const int blk = blockIdx.x;
  const int tt = blk & 63, bh = blk >> 6;
  const int b = bh >> 3, h = bh & 7;
  const float* qb = qr + ((size_t)bh * T_ + tt * 16) * 64;
  const float* kb = kr + (size_t)bh * S_ * 64;
  const float* vb = vt + (size_t)bh * 64 * S_;              // transposed [d][s]
  const int lane = threadIdx.x, l16 = lane & 15, lh = lane >> 4;
  __shared__ alignas(16) float sc[16][36];

  v16bf aq0 = cvt16(qb + l16 * 64 + lh * 16);
  v16bf aq1 = cvt16(qb + l16 * 64 + 32 + lh * 16);
  float mrow = 0.f, linv = 1.f;
  if (lane < 16) {
    mrow = mstat[(size_t)bh * T_ + tt * 16 + lane];
    linv = 1.0f / lstat[(size_t)bh * T_ + tt * 16 + lane];
  }
  v8f oacc[4];
  #pragma unroll
  for (int i = 0; i < 4; i++) oacc[i] = zero8();

  for (int s0 = 0; s0 < S_; s0 += 32) {
    #pragma unroll
    for (int sn = 0; sn < 2; sn++) {
      const float* krow = kb + (size_t)(s0 + sn * 16 + l16) * 64;
      v8f acc = zero8();
      acc = wmma_bf16(aq0, cvt16(krow + lh * 16), acc);
      acc = wmma_bf16(aq1, cvt16(krow + 32 + lh * 16), acc);
      #pragma unroll
      for (int r = 0; r < 8; r++) sc[r + lh * 8][sn * 16 + l16] = acc[r];
    }
    __syncthreads();
    if (lane < 16)
      for (int c = 0; c < 32; c++)
        sc[lane][c] = __expf(sc[lane][c] - mrow) * linv;
    __syncthreads();

    // coverage column sums (each lane owns one s column of the chunk)
    float csum = 0.f;
    for (int r = 0; r < 16; r++) csum += sc[r][lane];
    atomicAdd(&cov[(size_t)b * S_ + s0 + lane], csum);

    // P fragment: 16 contiguous LDS floats per lane
    v16bf ap = cvt16(&sc[l16][lh * 16]);
    #pragma unroll
    for (int dt = 0; dt < 4; dt++) {
      // V^T row is s-contiguous: 16 consecutive floats
      v16bf bv = cvt16(vb + (size_t)(dt * 16 + l16) * S_ + s0 + lh * 16);
      oacc[dt] = wmma_bf16(ap, bv, oacc[dt]);
    }
    __syncthreads();
  }

  #pragma unroll
  for (int dt = 0; dt < 4; dt++)
    #pragma unroll
    for (int r = 0; r < 8; r++) {
      int m = tt * 16 + r + (lh << 3);
      attn[((size_t)(b * T_ + m) * H_ + h) * 64 + dt * 16 + l16] = oacc[dt][r];
    }
}

__global__ __launch_bounds__(256)
void cov_final_k(const float* __restrict__ cov_in, const float* __restrict__ cov_acc,
                 float* __restrict__ out)
{
  int i = blockIdx.x * 256 + threadIdx.x;                   // B*S = 4096
  out[i] = cov_in[i] + cov_acc[i] * (1.0f / (H_ * T_));
}

// ---------------------------------------------------------------------------
extern "C" void kernel_launch(void* const* d_in, const int* in_sizes, int n_in,
                              void* d_out, int out_size, void* d_ws, size_t ws_size,
                              hipStream_t stream)
{
  const float* x         = (const float*)d_in[0];
  const float* memory    = (const float*)d_in[1];
  const float* coverage  = (const float*)d_in[2];
  const float* in_proj_w = (const float*)d_in[3];
  const float* conv_w    = (const float*)d_in[4];
  const float* conv_b    = (const float*)d_in[5];
  const float* dt_bias   = (const float*)d_in[6];
  const float* A_log     = (const float*)d_in[7];
  const float* Dp        = (const float*)d_in[8];
  const float* ssm_nw    = (const float*)d_in[9];
  const float* ssm_out_w = (const float*)d_in[10];
  const float* q_w       = (const float*)d_in[11];
  const float* kv_w      = (const float*)d_in[12];
  const float* out_w     = (const float*)d_in[13];
  const float* out_b     = (const float*)d_in[14];
  const float* ffn_w1    = (const float*)d_in[15];
  const float* ffn_b1    = (const float*)d_in[16];
  const float* ffn_w2    = (const float*)d_in[17];
  const float* ffn_b2    = (const float*)d_in[18];
  const float* ln1_w     = (const float*)d_in[19];
  const float* ln1_b     = (const float*)d_in[20];
  const float* ln2_w     = (const float*)d_in[21];
  const float* ln2_b     = (const float*)d_in[22];
  const float* ln3_w     = (const float*)d_in[23];
  const float* ln3_b     = (const float*)d_in[24];
  const int*   q_offset  = (const int*)d_in[25];

  float* ws   = (float*)d_ws;
  float* outx = (float*)d_out;                 // 4*1024*512
  float* outc = outx + (size_t)BT * DM;        // 4*1024 coverage

  hipMemsetAsync(ws + OFF_H, 0, (size_t)B_ * S_ * sizeof(float), stream);

  // ---- Mamba2 block ----
  layernorm_k<<<BT, 256, 0, stream>>>(x, ln1_w, ln1_b, ws + OFF_XN);
  gemm_bf16_wmma<<<dim3((DPROJ + 63) / 64, BT / 64), 256, 0, stream>>>(
      ws + OFF_XN, in_proj_w, nullptr, nullptr, ws + OFF_B, BT, DPROJ, DM, 0);
  conv_silu_k<<<(BT * CONVD) / 256, 256, 0, stream>>>(ws + OFF_B, conv_w, conv_b, ws + OFF_C);
  ssm_scan_k<<<B_ * SSH, 256, 0, stream>>>(ws + OFF_B, ws + OFF_C, dt_bias, A_log, Dp, ws + OFF_D);
  gate_rmsnorm_k<<<BT, 256, 0, stream>>>(ws + OFF_D, ws + OFF_B, ssm_nw);
  gemm_bf16_wmma<<<dim3(DM / 64, BT / 64), 256, 0, stream>>>(
      ws + OFF_D, ssm_out_w, nullptr, x, ws + OFF_E, BT, DM, DI, 0);   // +residual x -> x1

  // ---- Cross attention ----
  layernorm_k<<<BT, 256, 0, stream>>>(ws + OFF_E, ln2_w, ln2_b, ws + OFF_XN);
  gemm_bf16_wmma<<<dim3(DM / 64, BT / 64), 256, 0, stream>>>(
      ws + OFF_XN, q_w, nullptr, nullptr, ws + OFF_B, BT, DM, DM, 0);           // q_raw
  gemm_bf16_wmma<<<dim3((2 * DM) / 64, BT / 64), 256, 0, stream>>>(
      memory, kv_w, nullptr, nullptr, ws + OFF_B + 2097152, BT, 2 * DM, DM, 0); // kv_raw
  rope_q_k<<<(BT * DM) / 256, 256, 0, stream>>>(ws + OFF_B, ws + OFF_C, q_offset);
  rope_kv_k<<<(BT * DM) / 256, 256, 0, stream>>>(ws + OFF_B + 2097152,
                                                 ws + OFF_C + 2097152, ws + OFF_D);
  attn_pass1<<<B_ * H_ * (T_ / 16), 32, 0, stream>>>(
      ws + OFF_C, ws + OFF_C + 2097152, ws + OFF_G, ws + OFF_G + 32768);
  attn_pass2<<<B_ * H_ * (T_ / 16), 32, 0, stream>>>(
      ws + OFF_C, ws + OFF_C + 2097152, ws + OFF_D, ws + OFF_G, ws + OFF_G + 32768,
      ws + OFF_D + 2097152, ws + OFF_H);
  gemm_bf16_wmma<<<dim3(DM / 64, BT / 64), 256, 0, stream>>>(
      ws + OFF_D + 2097152, out_w, out_b, ws + OFF_E, ws + OFF_F, BT, DM, DM, 0); // x2

  // ---- FFN ----
  layernorm_k<<<BT, 256, 0, stream>>>(ws + OFF_F, ln3_w, ln3_b, ws + OFF_XN);
  gemm_bf16_wmma<<<dim3((4 * DM) / 64, BT / 64), 256, 0, stream>>>(
      ws + OFF_XN, ffn_w1, ffn_b1, nullptr, ws + OFF_B, BT, 4 * DM, DM, 1);     // GELU
  gemm_bf16_wmma<<<dim3(DM / 64, BT / 64), 256, 0, stream>>>(
      ws + OFF_B, ffn_w2, ffn_b2, ws + OFF_F, outx, BT, DM, 4 * DM, 0);

  cov_final_k<<<(B_ * S_) / 256, 256, 0, stream>>>(coverage, ws + OFF_H, outc);
}
